// DeformConv3d_31044023616350
// MI455X (gfx1250) — matrix operand order repaired and checked by reading, not compiled, not run on hardware
//
#include <hip/hip_runtime.h>
#include <hip/hip_bf16.h>
#include <cstdint>

// ---------------------------------------------------------------------------
// DeformConv pipeline for MI455X (gfx1250, wave32, WMMA, async global->LDS).
//  K1: 6 fused 5x5 convs (off1/2/4/5, z, m) as WMMA f16 GEMM  M=96,K=1600,N=32768
//      tap-major K (k = tap*64 + c); im2col fill = async b128 straight to LDS
//      from a zero-padded NHWC f16 source (no predication in the hot loop).
//  K2: padded-NHWC-f16 bilinear gather + cubic-z combine -> central f16 [pix][576]
//  K3: final conv == per-pixel GEMM via WMMA f16: M=64,K=576,N=32768
// ---------------------------------------------------------------------------

typedef __attribute__((ext_vector_type(16))) _Float16 v16h;
typedef __attribute__((ext_vector_type(8)))  _Float16 v8h;
typedef __attribute__((ext_vector_type(8)))  float    v8f;

union V16 { v16h v; v8h h[2]; };

static constexpr int kB   = 2;
static constexpr int kC   = 64;
static constexpr int kH   = 128;
static constexpr int kW   = 128;
static constexpr int kHW  = kH * kW;
static constexpr int kPix = kB * kHW;          // 32768
static constexpr int kOC1 = 96;                // 90 conv outputs padded to 96
static constexpr int kK1  = kC * 25;           // 1600
static constexpr int kK3  = kC * 9;            // 576
static constexpr int kOut = 64;
static constexpr int kSP  = 132;               // srcT padded spatial (ring of 2)
static constexpr int kIP  = 130;               // imgT padded spatial (ring of 1)

__device__ __forceinline__ float sigmoidf_(float x) {
    return 1.0f / (1.0f + __expf(-x));
}

// CDNA5 async global->LDS copy (ASYNCcnt-tracked), one b128 per lane.
__device__ __forceinline__ void async_b128(unsigned ldsOff, const _Float16* gp) {
    asm volatile("global_load_async_to_lds_b128 %0, %1, off"
                 :: "v"(ldsOff), "v"(gp) : "memory");
}
__device__ __forceinline__ void wait_async0() {
    asm volatile("s_wait_asynccnt 0" ::: "memory");
}

typedef __attribute__((address_space(3))) _Float16 lds_f16;

// ---------------------------------------------------------------------------
// Prep kernels
// ---------------------------------------------------------------------------
__global__ void pack_weights_kernel(
    const float* __restrict__ w1, const float* __restrict__ w2,
    const float* __restrict__ w4, const float* __restrict__ w5,
    const float* __restrict__ wz, const float* __restrict__ wm,
    const float* __restrict__ b1, const float* __restrict__ b2,
    const float* __restrict__ b4, const float* __restrict__ b5,
    const float* __restrict__ bz, const float* __restrict__ bm,
    _Float16* __restrict__ Wm, float* __restrict__ biasP)
{
    int i = blockIdx.x * 256 + threadIdx.x;
    if (i < kOC1 * kK1) {
        int oc = i / kK1, k = i % kK1;
        int tap = k >> 6;              // tap-major K ordering
        int c   = k & 63;
        const float* s = nullptr; int so = 0;
        if      (oc < 18) { s = w1; so = oc;      }
        else if (oc < 36) { s = w2; so = oc - 18; }
        else if (oc < 54) { s = w4; so = oc - 36; }
        else if (oc < 72) { s = w5; so = oc - 54; }
        else if (oc < 81) { s = wz; so = oc - 72; }
        else if (oc < 90) { s = wm; so = oc - 81; }
        float v = s ? s[so * kK1 + c * 25 + tap] : 0.0f;  // [oc][c][5][5]
        Wm[i] = (_Float16)v;
    }
    if (i < kOC1) {
        int oc = i;
        const float* s = nullptr; int so = 0;
        if      (oc < 18) { s = b1; so = oc;      }
        else if (oc < 36) { s = b2; so = oc - 18; }
        else if (oc < 54) { s = b4; so = oc - 36; }
        else if (oc < 72) { s = b5; so = oc - 54; }
        else if (oc < 81) { s = bz; so = oc - 72; }
        else if (oc < 90) { s = bm; so = oc - 81; }
        biasP[i] = s ? s[so] : 0.0f;
    }
}

__global__ void pack_wc_kernel(const float* __restrict__ w_conv,
                               _Float16* __restrict__ wc)
{
    // w_conv [64][64][3][3] flat == oc*576 + c*9 + (ky*3+kx): already GEMM layout
    int i = blockIdx.x * 256 + threadIdx.x;
    if (i < kOut * kK3) wc[i] = (_Float16)w_conv[i];
}

__global__ void transpose_imgs_pad_kernel(
    const float* __restrict__ i1, const float* __restrict__ i2,
    const float* __restrict__ i4, const float* __restrict__ i5,
    _Float16* __restrict__ imgTp)
{
    // NCHW f32 -> [img][b][130][130][c] f16, zero ring of 1 (padded frame coords)
    long i = (long)blockIdx.x * 256 + threadIdx.x;
    if (i >= (long)4 * kB * kIP * kIP * kC) return;
    int c = (int)(i & 63);
    long r = i >> 6;
    int y = (int)(r % kIP); r /= kIP;
    int x = (int)(r % kIP); r /= kIP;
    int b  = (int)(r & 1);
    int im = (int)(r >> 1);
    const float* s = (im == 0) ? i1 : (im == 1) ? i2 : (im == 2) ? i4 : i5;
    float v = 0.0f;
    if (x >= 1 && x <= kH && y >= 1 && y <= kW)
        v = s[((b * kC + c) << 14) + (x - 1) * kW + (y - 1)];
    imgTp[i] = (_Float16)v;
}

__global__ void transpose_src_pad_kernel(const float* __restrict__ src,
                                         _Float16* __restrict__ srcTp)
{
    // offset_source NCHW f32 -> [b][132][132][c] f16, zero ring of 2
    long i = (long)blockIdx.x * 256 + threadIdx.x;
    if (i >= (long)kB * kSP * kSP * kC) return;
    int c = (int)(i & 63);
    long r = i >> 6;
    int y = (int)(r % kSP); r /= kSP;
    int x = (int)(r % kSP);
    int b = (int)(r / kSP);
    float v = 0.0f;
    if (x >= 2 && x < kH + 2 && y >= 2 && y < kW + 2)
        v = src[((b * kC + c) << 14) + (x - 2) * kW + (y - 2)];
    srcTp[i] = (_Float16)v;
}

// ---------------------------------------------------------------------------
// K1: fused offset/z/m convolutions via WMMA + async LDS fill.
// grid: (b,h,wseg) = 2*128*2 = 512 blocks of 256 threads (8 waves).
// workgroup = 64 pixels x 96 oc; wave = 16 px x 48 oc (3 acc tiles).
// K chunk (32 wide) has constant tap -> fill = one async b128 per thread.
// ---------------------------------------------------------------------------
__global__ __launch_bounds__(256)
void conv_offsets_kernel(const _Float16* __restrict__ srcTp,
                         const _Float16* __restrict__ Wm,
                         const float* __restrict__ biasP,
                         float* __restrict__ off_all)
{
    __shared__ _Float16 Bs[2][64][40];  // double-buffered 64px x 32k (+8 pad)
    static constexpr int kBufB = 64 * 40 * 2;   // buffer stride (bytes)
    static constexpr int kNC   = kK1 / 32;      // 50 chunks

    int blk  = blockIdx.x;
    int wseg = blk & 1;
    int h    = (blk >> 1) & (kH - 1);
    int b    = blk >> 8;
    int w0   = wseg * 64;

    int tid  = threadIdx.x;
    int lane = tid & 31;
    int wave = tid >> 5;
    int pixTile = wave >> 1;
    int ocGrp   = wave & 1;
    int g  = lane >> 4;
    int nn = lane & 15;

    int fpi  = tid >> 2;              // fill: pixel 0..63
    int csub = (tid & 3) * 8;         // fill: 8 consecutive channels

    // thread-constant global base at (b, h+2, w0+fpi+2, csub) in padded src
    const _Float16* gbase =
        srcTp + ((((size_t)b * kSP + (h + 2)) * kSP) + (w0 + fpi + 2)) * (size_t)kC + csub;
    unsigned ldsBase = (unsigned)(uintptr_t)(lds_f16*)&Bs[0][fpi][csub];

    v8f acc[3] = {v8f{}, v8f{}, v8f{}};

    // issue chunk 0
    async_b128(ldsBase, gbase + ((0 / 5 - 2) * kSP + (0 % 5 - 2)) * kC);

    #pragma unroll 1
    for (int chunk = 0; chunk < kNC; ++chunk) {
        int buf = chunk & 1;
        wait_async0();                // own async stores to LDS landed
        __syncthreads();              // => everyone's stores for this chunk landed

        if (chunk + 1 < kNC) {
            int nc  = chunk + 1;
            int tap = nc >> 1;
            int dlt = ((tap / 5 - 2) * kSP + (tap % 5 - 2)) * kC + ((nc & 1) << 5);
            async_b128(ldsBase + (nc & 1) * kBufB, gbase + dlt);
        }

        // B fragment: col = lane%16, 16 consecutive halves at k = g*16
        V16 bf;
        bf.h[0] = *(const v8h*)&Bs[buf][pixTile * 16 + nn][g * 16];
        bf.h[1] = *(const v8h*)&Bs[buf][pixTile * 16 + nn][g * 16 + 8];

        #pragma unroll
        for (int t = 0; t < 3; ++t) {
            int oc = (ocGrp * 3 + t) * 16 + nn;       // A row = lane%16
            const _Float16* ar = Wm + (size_t)oc * kK1 + chunk * 32;
            V16 af;
            af.h[0] = *(const v8h*)(ar + g * 16);
            af.h[1] = *(const v8h*)(ar + g * 16 + 8);
            acc[t] = __builtin_amdgcn_wmma_f32_16x16x32_f16(
                false, af.v, false, bf.v, (short)0, acc[t], false, false);
        }
    }

    // epilogue: bias + activation, scatter to off_all[b][oc][h][w]
    int pw = w0 + pixTile * 16 + nn;
    #pragma unroll
    for (int t = 0; t < 3; ++t) {
        int ocT = ocGrp * 3 + t;
        #pragma unroll
        for (int gg = 0; gg < 8; ++gg) {
            int oc = ocT * 16 + g * 8 + gg;           // D: M = vgpr + (lane/16)*8
            if (oc >= 90) continue;
            float v = acc[t][gg] + biasP[oc];
            if      (oc >= 81) v = sigmoidf_(v);          // mask m
            else if (oc >= 72) v = 3.0f * sigmoidf_(v);   // z = 3*sigmoid
            off_all[((b * 90 + oc) * kH + h) * kW + pw] = v;
        }
    }
}

// ---------------------------------------------------------------------------
// K2: bilinear gather (padded NHWC f16, no bounds checks) + cubic-z combine.
// block = 4 pixels x 64 channels; corner offsets pre-scaled in LDS.
// ---------------------------------------------------------------------------
__global__ __launch_bounds__(256)
void deform_combine_kernel(const float* __restrict__ off_all,
                           const _Float16* __restrict__ imgTp,
                           _Float16* __restrict__ central)
{
    __shared__ int4   sCorn[4][9][4];   // {x_lt*130*64, y_lt*64, x_rb*130*64, y_rb*64}
    __shared__ float4 sGw[4][9][4];     // {g_lt, g_rb, g_lb, g_rt}
    __shared__ float  sZw[4][9][4];     // m * z-weight per image

    int tid  = threadIdx.x;
    int pix0 = blockIdx.x * 4;

    if (tid < 144) {
        int img = tid & 3;
        int n   = (tid >> 2) % 9;
        int p   = tid / 36;
        int pix = pix0 + p;
        int b = pix >> 14, rem = pix & (kHW - 1);
        int h = rem >> 7,  w   = rem & (kW - 1);
        int base = img * 18;
        float ox = off_all[((b * 90 + base + n) * kH + h) * kW + w];
        float oy = off_all[((b * 90 + base + 9 + n) * kH + h) * kW + w];
        float px = (float)(h + 1 + n / 3 - 1) + ox;   // padded-frame coords
        float py = (float)(w + 1 + n % 3 - 1) + oy;
        float flx = floorf(px), fly = floorf(py);
        float qltx = fminf(fmaxf(flx,        0.f), 129.f);
        float qrbx = fminf(fmaxf(flx + 1.f,  0.f), 129.f);
        float qlty = fminf(fmaxf(fly,        0.f), 129.f);
        float qrby = fminf(fmaxf(fly + 1.f,  0.f), 129.f);
        float pxc  = fminf(fmaxf(px, 0.f), 129.f);
        float pyc  = fminf(fmaxf(py, 0.f), 129.f);
        float glt = (1.f + (qltx - pxc)) * (1.f + (qlty - pyc));
        float grb = (1.f - (qrbx - pxc)) * (1.f - (qrby - pyc));
        float glb = (1.f - (qrbx - pxc)) * (1.f + (qlty - pyc));
        float grt = (1.f + (qltx - pxc)) * (1.f - (qrby - pyc));
        sCorn[p][n][img] = make_int4((int)qltx * kIP * kC, (int)qlty * kC,
                                     (int)qrbx * kIP * kC, (int)qrby * kC);
        sGw[p][n][img]   = make_float4(glt, grb, glb, grt);
    } else if (tid < 180) {
        int e = tid - 144;
        int p = e / 9, n = e % 9;
        int pix = pix0 + p;
        int b = pix >> 14, rem = pix & (kHW - 1);
        int h = rem >> 7,  w   = rem & (kW - 1);
        float z  = off_all[((b * 90 + 72 + n) * kH + h) * kW + w];
        float mm = off_all[((b * 90 + 81 + n) * kH + h) * kW + w];
        float z2 = z * z, z3 = z2 * z;
        sZw[p][n][0] = mm * (1.f - (11.f / 6.f) * z + z2 - z3 / 6.f);
        sZw[p][n][1] = mm * (3.f * z - 2.5f * z2 + 0.5f * z3);
        sZw[p][n][2] = mm * (-1.5f * z + 2.f * z2 - 0.5f * z3);
        sZw[p][n][3] = mm * (z / 3.f - 0.5f * z2 + z3 / 6.f);
    }
    __syncthreads();

    int c   = tid & 63;
    int p   = tid >> 6;
    int pix = pix0 + p;
    int b   = pix >> 14;

    const size_t plane = (size_t)kIP * kIP * kC;
    _Float16* out = central + (size_t)pix * kK3 + c * 9;

    #pragma unroll
    for (int n = 0; n < 9; ++n) {
        float acc = 0.f;
        #pragma unroll
        for (int img = 0; img < 4; ++img) {
            int4   q  = sCorn[p][n][img];
            float4 gw = sGw[p][n][img];
            const _Float16* ib = imgTp + (size_t)(img * kB + b) * plane + c;
            float vlt = (float)ib[q.x + q.y];
            float vrb = (float)ib[q.z + q.w];
            float vlb = (float)ib[q.z + q.y];
            float vrt = (float)ib[q.x + q.w];
            float s = gw.x * vlt + gw.y * vrb + gw.z * vlb + gw.w * vrt;
            acc += sZw[p][n][img] * s;
        }
        out[n] = (_Float16)acc;
    }
}

// ---------------------------------------------------------------------------
// K3: out[oc,pix] = Wc[64][576] x central^T via WMMA. 512 blocks, 8 waves;
// workgroup = 64 pixels x 64 oc; wave = 16 px x 32 oc (2 acc tiles).
// ---------------------------------------------------------------------------
__global__ __launch_bounds__(256)
void final_gemm_kernel(const _Float16* __restrict__ central,
                       const _Float16* __restrict__ wc,
                       float* __restrict__ out)
{
    int tid  = threadIdx.x;
    int lane = tid & 31;
    int wave = tid >> 5;
    int pixTile = wave >> 1;
    int ocGrp   = wave & 1;
    int g  = lane >> 4;
    int nn = lane & 15;

    int pix = blockIdx.x * 64 + pixTile * 16 + nn;
    const _Float16* crow = central + (size_t)pix * kK3;

    v8f acc[2] = {v8f{}, v8f{}};

    #pragma unroll 1
    for (int chunk = 0; chunk < kK3 / 32; ++chunk) {
        V16 bf;   // B col = lane%16 (pixel), 16 consecutive K halves
        bf.h[0] = *(const v8h*)(crow + chunk * 32 + g * 16);
        bf.h[1] = *(const v8h*)(crow + chunk * 32 + g * 16 + 8);
        #pragma unroll
        for (int t = 0; t < 2; ++t) {
            int oc = (ocGrp * 2 + t) * 16 + nn;
            const _Float16* ar = wc + (size_t)oc * kK3 + chunk * 32;
            V16 af;
            af.h[0] = *(const v8h*)(ar + g * 16);
            af.h[1] = *(const v8h*)(ar + g * 16 + 8);
            acc[t] = __builtin_amdgcn_wmma_f32_16x16x32_f16(
                false, af.v, false, bf.v, (short)0, acc[t], false, false);
        }
    }

    int b  = pix >> 14;
    int hw = pix & (kHW - 1);
    #pragma unroll
    for (int t = 0; t < 2; ++t) {
        #pragma unroll
        for (int gg = 0; gg < 8; ++gg) {
            int oc = (ocGrp * 2 + t) * 16 + g * 8 + gg;
            out[((size_t)b * kOut + oc) * kHW + hw] = acc[t][gg];
        }
    }
}

// ---------------------------------------------------------------------------
extern "C" void kernel_launch(void* const* d_in, const int* in_sizes, int n_in,
                              void* d_out, int out_size, void* d_ws, size_t ws_size,
                              hipStream_t stream)
{
    const float* img1 = (const float*)d_in[0];
    const float* img2 = (const float*)d_in[1];
    const float* img4 = (const float*)d_in[2];
    const float* img5 = (const float*)d_in[3];
    const float* offset_source = (const float*)d_in[4];
    const float* w_pc1 = (const float*)d_in[5];
    const float* b_pc1 = (const float*)d_in[6];
    const float* w_pc2 = (const float*)d_in[7];
    const float* b_pc2 = (const float*)d_in[8];
    const float* w_pc4 = (const float*)d_in[9];
    const float* b_pc4 = (const float*)d_in[10];
    const float* w_pc5 = (const float*)d_in[11];
    const float* b_pc5 = (const float*)d_in[12];
    const float* w_z   = (const float*)d_in[13];
    const float* b_z   = (const float*)d_in[14];
    const float* w_m   = (const float*)d_in[15];
    const float* b_m   = (const float*)d_in[16];
    const float* w_conv = (const float*)d_in[17];
    float* outp = (float*)d_out;

    auto al = [](size_t x) { return (x + 255) & ~(size_t)255; };
    char* ws = (char*)d_ws;
    size_t o = 0;
    float*    off_all = (float*)(ws + o);    o += al((size_t)kB * 90 * kHW * sizeof(float));
    _Float16* central = (_Float16*)(ws + o); o += al((size_t)kPix * kK3 * sizeof(_Float16));
    _Float16* Wm      = (_Float16*)(ws + o); o += al((size_t)kOC1 * kK1 * sizeof(_Float16));
    float*    biasP   = (float*)(ws + o);    o += al((size_t)kOC1 * sizeof(float));
    _Float16* wcP     = (_Float16*)(ws + o); o += al((size_t)kOut * kK3 * sizeof(_Float16));
    _Float16* imgTp   = (_Float16*)(ws + o); o += al((size_t)4 * kB * kIP * kIP * kC * sizeof(_Float16));
    _Float16* srcTp   = (_Float16*)(ws + o); o += al((size_t)kB * kSP * kSP * kC * sizeof(_Float16));

    pack_weights_kernel<<<(kOC1 * kK1 + 255) / 256, 256, 0, stream>>>(
        w_pc1, w_pc2, w_pc4, w_pc5, w_z, w_m,
        b_pc1, b_pc2, b_pc4, b_pc5, b_z, b_m, Wm, biasP);
    pack_wc_kernel<<<(kOut * kK3 + 255) / 256, 256, 0, stream>>>(w_conv, wcP);

    long nImgP = (long)4 * kB * kIP * kIP * kC;
    transpose_imgs_pad_kernel<<<(int)((nImgP + 255) / 256), 256, 0, stream>>>(
        img1, img2, img4, img5, imgTp);
    long nSrcP = (long)kB * kSP * kSP * kC;
    transpose_src_pad_kernel<<<(int)((nSrcP + 255) / 256), 256, 0, stream>>>(
        offset_source, srcTp);

    conv_offsets_kernel<<<kB * kH * (kW / 64), 256, 0, stream>>>(
        srcTp, Wm, biasP, off_all);

    deform_combine_kernel<<<kPix / 4, 256, 0, stream>>>(off_all, imgTp, central);

    final_gemm_kernel<<<kPix / 64, 256, 0, stream>>>(central, wcP, outp);
}
